// ROIHead_91268055040557
// MI455X (gfx1250) — compile-verified, matrix operand order checked by reading, not run on hardware
//
#include <hip/hip_runtime.h>
#include <hip/hip_bf16.h>
#include <math.h>

typedef __attribute__((ext_vector_type(16))) __bf16 v16bf;
typedef __attribute__((ext_vector_type(8)))  float  v8f;

#define POOL    7
#define NCLS    91
#define NPROP   2000
#define FH      50
#define FW      50
#define FC      256
#define D_IN    12544      // 256*7*7
#define H1      1024
#define NH_REAL 455        // 91 cls + 364 reg
#define NH_PAD  464        // padded to multiple of 16
#define NTOPK   100
#define DW_MAXF 4.135166556742356f

static __device__ __forceinline__ unsigned short f2bf(float f) {
  unsigned int u = __float_as_uint(f);
  u += 0x7FFFu + ((u >> 16) & 1u);     // round-to-nearest-even
  return (unsigned short)(u >> 16);
}

static __device__ __forceinline__ unsigned int lds_addr32(const void* p) {
  // generic pointer to __shared__: low 32 bits are the wave-relative LDS offset
  return (unsigned int)(unsigned long long)p;
}

// ---------------- fp32 -> bf16 conversion ----------------
__global__ void f32_to_bf16_kernel(const float* __restrict__ src,
                                   unsigned short* __restrict__ dst, int n) {
  int i = blockIdx.x * blockDim.x + threadIdx.x;
  if (i < n) dst[i] = f2bf(src[i]);
}

// pack cls_w (91x1024) + reg_w (364x1024) into padded 464x1024 bf16
__global__ void head_pack_w_kernel(const float* __restrict__ cls_w,
                                   const float* __restrict__ reg_w,
                                   unsigned short* __restrict__ dst) {
  int i = blockIdx.x * blockDim.x + threadIdx.x;
  if (i >= NH_PAD * H1) return;
  int row = i / H1, col = i - row * H1;
  float v = 0.f;
  if (row < NCLS)          v = cls_w[row * H1 + col];
  else if (row < NH_REAL)  v = reg_w[(row - NCLS) * H1 + col];
  dst[i] = f2bf(v);
}

__global__ void head_pack_b_kernel(const float* __restrict__ cls_b,
                                   const float* __restrict__ reg_b,
                                   float* __restrict__ dst) {
  int i = blockIdx.x * blockDim.x + threadIdx.x;
  if (i >= NH_PAD) return;
  dst[i] = (i < NCLS) ? cls_b[i] : ((i < NH_REAL) ? reg_b[i - NCLS] : 0.f);
}

// ---------------- ROI max-pool, bf16 output ----------------
__global__ void roi_pool_kernel(const float* __restrict__ feat,
                                const float* __restrict__ rois,
                                const int* __restrict__ img_h,
                                const int* __restrict__ img_w,
                                unsigned short* __restrict__ pooled) {
  int idx = blockIdx.x * blockDim.x + threadIdx.x;
  if (idx >= NPROP * D_IN) return;
  int n = idx / D_IN;
  int d = idx - n * D_IN;
  int c = d / (POOL * POOL);
  int b = d - c * (POOL * POOL);
  int ph = b / POOL, pw = b - ph * POOL;

  float scale = fminf((float)FH / (float)img_h[0], (float)FW / (float)img_w[0]);
  float x1 = rintf(rois[n * 4 + 0] * scale);
  float y1 = rintf(rois[n * 4 + 1] * scale);
  float x2 = rintf(rois[n * 4 + 2] * scale);
  float y2 = rintf(rois[n * 4 + 3] * scale);
  float rw = fmaxf(x2 - x1 + 1.f, 1.f);
  float rh = fmaxf(y2 - y1 + 1.f, 1.f);

  int hs = (int)fminf(fmaxf(floorf((float)ph * rh / POOL) + y1, 0.f), (float)FH);
  int he = (int)fminf(fmaxf(ceilf((float)(ph + 1) * rh / POOL) + y1, 0.f), (float)FH);
  int wsx = (int)fminf(fmaxf(floorf((float)pw * rw / POOL) + x1, 0.f), (float)FW);
  int wex = (int)fminf(fmaxf(ceilf((float)(pw + 1) * rw / POOL) + x1, 0.f), (float)FW);

  float m = -1e30f;
  const float* fc = feat + (size_t)c * FH * FW;
  for (int h = hs; h < he; ++h)
    for (int w = wsx; w < wex; ++w)
      m = fmaxf(m, fc[h * FW + w]);
  if (hs >= he || wsx >= wex) m = 0.f;
  pooled[(size_t)n * D_IN + d] = f2bf(m);
}

// ======================================================================
// FC GEMM with async-LDS staged B panel.
//   C[m][n] = relu(sum_k A[m][k]*B[n][k] + bias[n]),  bf16 in, bf16 out.
// Block = 256 threads = 8 waves. Block tile: 128 M-rows x 64 N-cols.
// All 8 waves share one 64x32 B K-slice staged in LDS (double buffered)
// via GLOBAL_LOAD_ASYNC_TO_LDS_B128 (ASYNCcnt), cutting L2 B-traffic 8x.
// Requires: N % 64 == 0, K % 32 == 0, M % 16 == 0.
// ======================================================================
__global__ void __launch_bounds__(256)
gemm_fc_wmma_async(const unsigned short* __restrict__ A,
                   const unsigned short* __restrict__ B,
                   const float* __restrict__ bias,
                   unsigned short* __restrict__ Cb,
                   int M, int N, int K) {
  __shared__ unsigned short sB[2][64 * 32];       // 2 x 4KB: [col 0..63][k 0..31]

  const int ntn  = N >> 6;                        // N/64 column groups
  const int tn   = blockIdx.x % ntn;
  const int tmg  = blockIdx.x / ntn;              // group of 8 M-tiles
  const int wave = threadIdx.x >> 5;
  const int lane = threadIdx.x & 31;
  const int half = lane >> 4;                     // K-half split (ISA 16-bit A layout)
  const int r    = lane & 15;                     // row (A) / col (B) within 16-tile
  const int tm   = tmg * 8 + wave;
  const bool active = (tm * 16) < M;              // wave-uniform

  // ---- async B fill: thread i copies one 16B chunk of the 64x32 tile ----
  const int fcol  = threadIdx.x >> 2;             // 0..63
  const int fpart = threadIdx.x & 3;              // 0..3 (k-chunk of 8)
  const unsigned short* Bbase = B + (size_t)(tn * 64) * K;          // uniform
  const unsigned int goff0 = (unsigned int)((fcol * K + fpart * 8) * 2);
  const unsigned int ldsA[2] = {
      lds_addr32(&sB[0][fcol * 32 + fpart * 8]),
      lds_addr32(&sB[1][fcol * 32 + fpart * 8])};

  const unsigned short* Arow = A + (size_t)(tm * 16 + r) * K;

  const v8f vzero = {0.f, 0.f, 0.f, 0.f, 0.f, 0.f, 0.f, 0.f};
  v8f acc[4] = {vzero, vzero, vzero, vzero};

  union U { uint4 q[2]; v16bf v; };

  // preload K-slice 0 into buffer 0
  asm volatile("global_load_async_to_lds_b128 %0, %1, %2"
               :: "v"(ldsA[0]), "v"(goff0), "s"(Bbase) : "memory");
  asm volatile("s_wait_asynccnt 0x0" ::: "memory");
  __syncthreads();

  int buf = 0;
  for (int kb = 0; kb < K; kb += 32) {
    const int nkb = kb + 32;
    if (nkb < K) {  // issue next slice into the other buffer (overlaps WMMA)
      asm volatile("global_load_async_to_lds_b128 %0, %1, %2"
                   :: "v"(ldsA[buf ^ 1]), "v"(goff0 + (unsigned int)nkb * 2),
                      "s"(Bbase) : "memory");
    }
    if (active) {
      __builtin_prefetch(Arow + kb + 256, 0, 1);  // global_prefetch_b8
      U ua;
      ua.q[0] = *(const uint4*)(Arow + kb + half * 8);       // K = half*8..+8
      ua.q[1] = *(const uint4*)(Arow + kb + 16 + half * 8);  // K = 16+half*8..+8
      const unsigned short* bb = &sB[buf][0];
#pragma unroll
      for (int t = 0; t < 4; ++t) {                          // ds_load_b128 x2
        U ub;
        ub.q[0] = *(const uint4*)(bb + (t * 16 + r) * 32 + half * 8);
        ub.q[1] = *(const uint4*)(bb + (t * 16 + r) * 32 + 16 + half * 8);
        acc[t] = __builtin_amdgcn_wmma_f32_16x16x32_bf16(
            false, ua.v, false, ub.v, (short)0, acc[t], false, false);
      }
    }
    __syncthreads();                  // all waves done reading sB[buf]
    if (nkb < K) {
      asm volatile("s_wait_asynccnt 0x0" ::: "memory");  // my fill complete
      __syncthreads();                // everyone's fill complete & visible
      buf ^= 1;
    }
  }

  if (active) {
#pragma unroll
    for (int t = 0; t < 4; ++t) {
      const int col = tn * 64 + t * 16 + r;
      const float bb = bias ? bias[col] : 0.f;
#pragma unroll
      for (int v = 0; v < 8; ++v) {
        // D layout: vgpr v -> row v (lanes 0-15) / v+8 (lanes 16-31)
        int row = tm * 16 + v + half * 8;
        float val = fmaxf(acc[t][v] + bb, 0.f);   // relu
        Cb[(size_t)row * N + col] = f2bf(val);
      }
    }
  }
}

// ---------------- direct-load WMMA GEMM (head layer, fp32 out) ----------------
__global__ void gemm_head_wmma(const unsigned short* __restrict__ A,
                               const unsigned short* __restrict__ B,
                               const float* __restrict__ bias,
                               float* __restrict__ Cf,
                               int M, int N, int K) {
  const int wid     = blockIdx.x * (blockDim.x >> 5) + (threadIdx.x >> 5);
  const int waves_n = N >> 4;
  const int total   = (M >> 4) * waves_n;
  if (wid >= total) return;                 // wave-uniform; EXEC all-1s for WMMA
  const int tm   = wid / waves_n;
  const int tn   = wid % waves_n;
  const int lane = threadIdx.x & 31;
  const int half = lane >> 4;
  const int r    = lane & 15;

  const unsigned short* Arow = A + (size_t)(tm * 16 + r) * K;
  const unsigned short* Brow = B + (size_t)(tn * 16 + r) * K;

  union U { uint4 q[2]; v16bf v; };
  const v8f vzero = {0.f, 0.f, 0.f, 0.f, 0.f, 0.f, 0.f, 0.f};
  v8f acc = vzero;

  for (int kb = 0; kb < K; kb += 32) {
    U ua, ub;
    ua.q[0] = *(const uint4*)(Arow + kb + half * 8);
    ua.q[1] = *(const uint4*)(Arow + kb + 16 + half * 8);
    ub.q[0] = *(const uint4*)(Brow + kb + half * 8);
    ub.q[1] = *(const uint4*)(Brow + kb + 16 + half * 8);
    acc = __builtin_amdgcn_wmma_f32_16x16x32_bf16(
        false, ua.v, false, ub.v, (short)0, acc, false, false);
  }

  const int col = tn * 16 + r;
  const float bb = bias ? bias[col] : 0.f;
#pragma unroll
  for (int v = 0; v < 8; ++v) {
    int row = tm * 16 + v + half * 8;
    Cf[(size_t)row * N + col] = acc[v] + bb;
  }
}

// ---------------- softmax + box decode ----------------
__global__ void decode_kernel(const float* __restrict__ head,
                              const float* __restrict__ props,
                              const int* __restrict__ img_h,
                              const int* __restrict__ img_w,
                              float* __restrict__ boxes,
                              float* __restrict__ scores,
                              int* __restrict__ labels,
                              int* __restrict__ valid) {
  int n = blockIdx.x * blockDim.x + threadIdx.x;
  if (n >= NPROP) return;
  const float* row = head + (size_t)n * NH_PAD;

  float mx = row[0];
  for (int c = 1; c < NCLS; ++c) mx = fmaxf(mx, row[c]);
  float sum = 0.f;
  for (int c = 0; c < NCLS; ++c) sum += expf(row[c] - mx);
  float best = -1.f; int bl = 1;
  for (int c = 1; c < NCLS; ++c) {
    float p = expf(row[c] - mx) / sum;
    if (p > best) { best = p; bl = c; }
  }

  float dx = row[NCLS + bl * 4 + 0];
  float dy = row[NCLS + bl * 4 + 1];
  float dw = fminf(row[NCLS + bl * 4 + 2], DW_MAXF);
  float dh = fminf(row[NCLS + bl * 4 + 3], DW_MAXF);

  float p0 = props[n * 4 + 0], p1 = props[n * 4 + 1];
  float p2 = props[n * 4 + 2], p3 = props[n * 4 + 3];
  float pw = p2 - p0, ph = p3 - p1;
  float pcx = p0 + 0.5f * pw, pcy = p1 + 0.5f * ph;
  float cx = dx * pw + pcx, cy = dy * ph + pcy;
  float bw = expf(dw) * pw, bh = expf(dh) * ph;
  float W = (float)img_w[0], H = (float)img_h[0];
  float b0 = fminf(fmaxf(cx - 0.5f * bw, 0.f), W);
  float b1 = fminf(fmaxf(cy - 0.5f * bh, 0.f), H);
  float b2 = fminf(fmaxf(cx + 0.5f * bw, 0.f), W);
  float b3 = fminf(fmaxf(cy + 0.5f * bh, 0.f), H);

  boxes[n * 4 + 0] = b0; boxes[n * 4 + 1] = b1;
  boxes[n * 4 + 2] = b2; boxes[n * 4 + 3] = b3;
  scores[n] = best;
  labels[n] = bl;
  valid[n]  = (best >= 0.05f) && ((b2 - b0) >= 1.f) && ((b3 - b1) >= 1.f);
}

// ---------------- one-block bitonic sort by (valid ? -score : +inf) ----------------
__global__ void sort_kernel(const float* __restrict__ scores,
                            const int* __restrict__ valid,
                            int* __restrict__ order) {
  __shared__ float skey[2048];
  __shared__ int   sidx[2048];
  int tid = threadIdx.x;
  for (int i = tid; i < 2048; i += blockDim.x) {
    if (i < NPROP) { skey[i] = valid[i] ? -scores[i] : INFINITY; sidx[i] = i; }
    else           { skey[i] = INFINITY; sidx[i] = 0x7fffffff; }
  }
  __syncthreads();
  for (int k = 2; k <= 2048; k <<= 1) {
    for (int j = k >> 1; j > 0; j >>= 1) {
      for (int i = tid; i < 2048; i += blockDim.x) {
        int ixj = i ^ j;
        if (ixj > i) {
          bool asc = ((i & k) == 0);
          float a = skey[i], b = skey[ixj];
          if (asc ? (a > b) : (a < b)) {
            skey[i] = b; skey[ixj] = a;
            int t = sidx[i]; sidx[i] = sidx[ixj]; sidx[ixj] = t;
          }
        }
      }
      __syncthreads();
    }
  }
  for (int i = tid; i < NPROP; i += blockDim.x) order[i] = sidx[i];
}

// ---------------- greedy class-wise NMS + top-100 output (one workgroup) -------------
__global__ void nms_out_kernel(const float* __restrict__ boxes,
                               const float* __restrict__ scores,
                               const int* __restrict__ labels,
                               const int* __restrict__ valid,
                               const int* __restrict__ order,
                               float* __restrict__ out) {
  __shared__ float bx[NPROP][4];
  __shared__ float sc[NPROP];
  __shared__ unsigned char lb[NPROP];
  __shared__ unsigned char vz[NPROP];
  __shared__ unsigned char sp[NPROP];
  __shared__ unsigned char kp[NPROP];
  int tid = threadIdx.x;

  for (int s = tid; s < NPROP; s += blockDim.x) {
    int o = order[s];
    if (o >= 0 && o < NPROP) {
      bx[s][0] = boxes[o * 4 + 0]; bx[s][1] = boxes[o * 4 + 1];
      bx[s][2] = boxes[o * 4 + 2]; bx[s][3] = boxes[o * 4 + 3];
      sc[s] = scores[o];
      lb[s] = (unsigned char)labels[o];
      vz[s] = valid[o] ? 1 : 0;
    } else {
      bx[s][0] = bx[s][1] = bx[s][2] = bx[s][3] = 0.f;
      sc[s] = 0.f; lb[s] = 0; vz[s] = 0;
    }
    sp[s] = 0; kp[s] = 0;
  }

  for (int i = 0; i < NPROP; ++i) {
    __syncthreads();
    bool keep_i = vz[i] && !sp[i];
    if (tid == 0) kp[i] = keep_i ? 1 : 0;
    if (keep_i) {
      float a0 = bx[i][0], a1 = bx[i][1], a2 = bx[i][2], a3 = bx[i][3];
      float areaA = (a2 - a0) * (a3 - a1);
      unsigned char li = lb[i];
      for (int j = i + 1 + tid; j < NPROP; j += blockDim.x) {
        if (lb[j] != li) continue;
        float b0 = bx[j][0], b1 = bx[j][1], b2 = bx[j][2], b3 = bx[j][3];
        float xl = fmaxf(a0, b0), yt = fmaxf(a1, b1);
        float xr = fminf(a2, b2), yb = fminf(a3, b3);
        float inter = fmaxf(xr - xl, 0.f) * fmaxf(yb - yt, 0.f);
        float areaB = (b2 - b0) * (b3 - b1);
        float iou = inter / (areaA + areaB - inter + 1e-9f);
        if (iou > 0.5f) sp[j] = 1;
      }
    }
  }
  __syncthreads();

  if (tid == 0) {
    int cnt = 0;
    for (int s = 0; s < NPROP && cnt < NTOPK; ++s) {
      if (kp[s]) {
        out[cnt * 4 + 0] = bx[s][0]; out[cnt * 4 + 1] = bx[s][1];
        out[cnt * 4 + 2] = bx[s][2]; out[cnt * 4 + 3] = bx[s][3];
        out[4 * NTOPK + cnt] = (float)lb[s];
        out[5 * NTOPK + cnt] = sc[s];
        ++cnt;
      }
    }
    for (; cnt < NTOPK; ++cnt) {
      out[cnt * 4 + 0] = 0.f; out[cnt * 4 + 1] = 0.f;
      out[cnt * 4 + 2] = 0.f; out[cnt * 4 + 3] = 0.f;
      out[4 * NTOPK + cnt] = 0.f;
      out[5 * NTOPK + cnt] = 0.f;
    }
  }
}

// ---------------- launcher ----------------
extern "C" void kernel_launch(void* const* d_in, const int* in_sizes, int n_in,
                              void* d_out, int out_size, void* d_ws, size_t ws_size,
                              hipStream_t stream) {
  const float* feat  = (const float*)d_in[0];
  const float* props = (const float*)d_in[1];
  const float* fc1_w = (const float*)d_in[2];
  const float* fc1_b = (const float*)d_in[3];
  const float* fc2_w = (const float*)d_in[4];
  const float* fc2_b = (const float*)d_in[5];
  const float* cls_w = (const float*)d_in[6];
  const float* cls_b = (const float*)d_in[7];
  const float* reg_w = (const float*)d_in[8];
  const float* reg_b = (const float*)d_in[9];
  const int*   img_h = (const int*)d_in[10];
  const int*   img_w = (const int*)d_in[11];
  float* out = (float*)d_out;

  // workspace carve-out (~91 MB total)
  char* wsb = (char*)d_ws;
  size_t off = 0;
  auto carve = [&](size_t bytes) -> char* {
    char* p = wsb + off;
    off = (off + bytes + 255) & ~(size_t)255;
    return p;
  };
  unsigned short* pooled = (unsigned short*)carve((size_t)NPROP * D_IN * 2);
  unsigned short* w1     = (unsigned short*)carve((size_t)H1 * D_IN * 2);
  unsigned short* w2     = (unsigned short*)carve((size_t)H1 * H1 * 2);
  unsigned short* wh     = (unsigned short*)carve((size_t)NH_PAD * H1 * 2);
  float*          bh     = (float*)carve((size_t)NH_PAD * 4);
  unsigned short* x1b    = (unsigned short*)carve((size_t)NPROP * H1 * 2);
  unsigned short* x2b    = (unsigned short*)carve((size_t)NPROP * H1 * 2);
  float*          hout   = (float*)carve((size_t)NPROP * NH_PAD * 4);
  float*          boxes  = (float*)carve((size_t)NPROP * 4 * 4);
  float*          scores = (float*)carve((size_t)NPROP * 4);
  int*            labels = (int*)carve((size_t)NPROP * 4);
  int*            valid  = (int*)carve((size_t)NPROP * 4);
  int*            order  = (int*)carve((size_t)NPROP * 4);

  // 1) weight conversion to bf16 (weights fit in the 192MB L2 afterwards)
  {
    int n1 = H1 * D_IN;
    f32_to_bf16_kernel<<<(n1 + 255) / 256, 256, 0, stream>>>(fc1_w, w1, n1);
    int n2 = H1 * H1;
    f32_to_bf16_kernel<<<(n2 + 255) / 256, 256, 0, stream>>>(fc2_w, w2, n2);
    int n3 = NH_PAD * H1;
    head_pack_w_kernel<<<(n3 + 255) / 256, 256, 0, stream>>>(cls_w, reg_w, wh);
    head_pack_b_kernel<<<(NH_PAD + 255) / 256, 256, 0, stream>>>(cls_b, reg_b, bh);
  }

  // 2) ROI max-pool -> bf16 activations
  {
    long long total = (long long)NPROP * D_IN;
    roi_pool_kernel<<<(int)((total + 255) / 256), 256, 0, stream>>>(
        feat, props, img_h, img_w, pooled);
  }

  // 3) FC1 + relu: (2000 x 12544) x (1024 x 12544)^T, async-LDS staged B
  {
    int tmg = (NPROP / 16 + 7) / 8;          // 16 groups of 8 M-tiles
    int grid = tmg * (H1 / 64);              // 16 * 16 = 256 blocks
    gemm_fc_wmma_async<<<grid, 256, 0, stream>>>(
        pooled, w1, fc1_b, x1b, NPROP, H1, D_IN);
  }
  // 4) FC2 + relu: (2000 x 1024) x (1024 x 1024)^T
  {
    int tmg = (NPROP / 16 + 7) / 8;
    int grid = tmg * (H1 / 64);
    gemm_fc_wmma_async<<<grid, 256, 0, stream>>>(
        x1b, w2, fc2_b, x2b, NPROP, H1, H1);
  }
  // 5) head: (2000 x 1024) x (464 x 1024)^T, fp32 out
  {
    int waves = (NPROP / 16) * (NH_PAD / 16);   // 125 * 29 = 3625
    gemm_head_wmma<<<(waves * 32 + 255) / 256, 256, 0, stream>>>(
        x2b, wh, bh, hout, NPROP, NH_PAD, H1);
  }

  // 6) softmax + box decode
  decode_kernel<<<(NPROP + 255) / 256, 256, 0, stream>>>(
      hout, props, img_h, img_w, boxes, scores, labels, valid);

  // 7) sort by descending score (invalid last)
  sort_kernel<<<1, 1024, 0, stream>>>(scores, valid, order);

  // 8) greedy NMS + top-100 write-out
  nms_out_kernel<<<1, 1024, 0, stream>>>(boxes, scores, labels, valid, order, out);
}
